// Dcn_apply_27358941676273
// MI455X (gfx1250) — compile-verified, matrix operand order checked by reading, not run on hardware
//
#include <hip/hip_runtime.h>
#include <hip/hip_bf16.h>

// ---------------------------------------------------------------------------
// Deformable conv v1 forward for MI455X (gfx1250, wave32, WMMA).
//   B=4, Cin=256, Cout=256, H=W=64, K=3, DG=4
//   Main contraction = GEMM  M=16384 (b*h*w), N=256 (Cout), K=2304 (g,c,k)
//   -> V_WMMA_F32_16X16X32_F16, f32 accumulate.
// ---------------------------------------------------------------------------

typedef __attribute__((ext_vector_type(16))) _Float16 v16h;
typedef __attribute__((ext_vector_type(8)))  _Float16 v8h;
typedef __attribute__((ext_vector_type(8)))  float    v8f;

#define BB   4
#define CIN  256
#define COUT 256
#define HH   64
#define WW   64
#define KK9  9
#define DG   4
#define CG   64            // Cin / DG
#define KDIM 2304          // DG * CG * KK9
#define MDIM 16384         // B*H*W
#define HW   4096          // H*W
#define KT   72            // KDIM / 32
#define LDA  40            // LDS A row stride in halves (32 + 8 pad, bank-conflict free)

// ---------------------------------------------------------------------------
// Kernel 0: pack w_deform (f32 [Cout][Cin][3][3]) into WMMA B-fragment layout
// (f16). Fragment for (n_tile nt, k_tile kt): 32 lanes x 16 halves, each lane
// holds column n = nt*16 + (lane&15); half j maps to K per ISA 16-bit B split:
//   lane<16 : K = kt*32 + (j<8 ? j   : j+8)   // K 0-7, 16-23
//   lane>=16: K = kt*32 + (j<8 ? j+8 : j+16)  // K 8-15, 24-31
// K index decomposes as kidx = g*576 + k*64 + c.
// ---------------------------------------------------------------------------
__global__ __launch_bounds__(256) void dcn_pack_w(
    const float* __restrict__ wd, _Float16* __restrict__ wpack)
{
    int idx = blockIdx.x * 256 + threadIdx.x;          // 16*72*32*16 = 589824
    if (idx >= 16 * KT * 32 * 16) return;
    int j    = idx & 15;
    int lane = (idx >> 4) & 31;
    int kt   = (idx >> 9) % KT;
    int nt   = idx / (KT * 32 * 16);
    int kin  = (lane < 16) ? ((j < 8) ? j : j + 8) : ((j < 8) ? j + 8 : j + 16);
    int kidx = kt * 32 + kin;
    int n    = nt * 16 + (lane & 15);
    int g    = kidx / 576;
    int rem  = kidx % 576;
    int k    = rem / CG;          // 0..8 == ky*3+kx
    int c    = rem % CG;
    int cin  = g * CG + c;
    wpack[idx] = (_Float16)wd[(n * CIN + cin) * KK9 + k];
}

// ---------------------------------------------------------------------------
// Kernel 1: offset matvec + bilinear im2col -> col[m][g*576 + k*64 + c] (f16)
// One thread per (m, g, k); inner loop over the 64 channels of the deform
// group, vectorized 16-byte stores (8 halves).
// ---------------------------------------------------------------------------
__global__ __launch_bounds__(256) void dcn_im2col(
    const float* __restrict__ x, const float* __restrict__ shape,
    const float* __restrict__ w_off, _Float16* __restrict__ col)
{
    int t = blockIdx.x * 256 + threadIdx.x;            // 16384*36 = 589824
    if (t >= MDIM * (DG * KK9)) return;
    int m  = t / 36;
    int gk = t % 36;                                   // g*9 + k
    int g  = gk / KK9;
    int k  = gk % KK9;
    int b  = m >> 12;
    int hw = m & (HW - 1);
    int h  = hw >> 6;
    int w  = hw & 63;

    // offset = shape(b,:,h,w) . w_offset(2*gk + {0,1}, :)
    const float* shp = shape + (size_t)b * 4 * HW + hw;
    float s0 = shp[0], s1 = shp[HW], s2 = shp[2 * HW], s3 = shp[3 * HW];
    const float* wo = w_off + gk * 8;                  // [72][4], dy row then dx row
    float dy = s0 * wo[0] + s1 * wo[1] + s2 * wo[2] + s3 * wo[3];
    float dx = s0 * wo[4] + s1 * wo[5] + s2 * wo[6] + s3 * wo[7];

    float ys = (float)(h + (k / 3) - 1) + dy;
    float xs = (float)(w + (k % 3) - 1) + dx;
    float y0f = floorf(ys), x0f = floorf(xs);
    float wy1 = ys - y0f,   wx1 = xs - x0f;
    int   y0  = (int)y0f,   x0  = (int)x0f;

    float cw[4];
    int   ci[4];
    {
        int yy[2] = { y0, y0 + 1 };
        int xx[2] = { x0, x0 + 1 };
        float fy[2] = { 1.0f - wy1, wy1 };
        float fx[2] = { 1.0f - wx1, wx1 };
#pragma unroll
        for (int a = 0; a < 2; ++a)
#pragma unroll
            for (int bb2 = 0; bb2 < 2; ++bb2) {
                int yc = yy[a], xc = xx[bb2];
                bool inb = (yc >= 0) & (yc <= HH - 1) & (xc >= 0) & (xc <= WW - 1);
                int ycl = min(max(yc, 0), HH - 1);
                int xcl = min(max(xc, 0), WW - 1);
                cw[a * 2 + bb2] = inb ? fy[a] * fx[bb2] : 0.0f;
                ci[a * 2 + bb2] = ycl * WW + xcl;
            }
    }

    const float* xb = x + ((size_t)b * CIN + g * CG) * HW;
    _Float16* dst = col + (size_t)m * KDIM + g * 576 + k * CG;

#pragma unroll
    for (int c0 = 0; c0 < CG; c0 += 8) {
        union { _Float16 hbuf[8]; uint4 u; } pk;
#pragma unroll
        for (int j = 0; j < 8; ++j) {
            const float* xc = xb + (size_t)(c0 + j) * HW;
            float v = cw[0] * xc[ci[0]] + cw[1] * xc[ci[1]]
                    + cw[2] * xc[ci[2]] + cw[3] * xc[ci[3]];
            pk.hbuf[j] = (_Float16)v;
        }
        *(uint4*)(dst + c0) = pk.u;                    // 16B coalesced store
    }
}

// ---------------------------------------------------------------------------
// Kernel 2: WMMA GEMM + ReLU.
//   grid = (M/128, N/64), block = 256 threads = 8 waves (4 M-subtiles x 2 N)
//   each wave: 32x32 output = 2x2 fragments, K-loop 72 x 32 with
//   double-buffered LDS staging of the A tile (128x32 f16).
// ---------------------------------------------------------------------------
__global__ __launch_bounds__(256) void dcn_wmma_gemm(
    const _Float16* __restrict__ col, const _Float16* __restrict__ wpack,
    float* __restrict__ out)
{
    __shared__ _Float16 ldsA[2][128 * LDA];

    const int lane = threadIdx.x & 31;
    const int wave = threadIdx.x >> 5;
    const int wy   = wave & 3;        // M sub-tile (x32)
    const int wx   = wave >> 2;       // N sub-tile (x32)
    const int m_base = blockIdx.x * 128;
    const int n_base = blockIdx.y * 64 + wx * 32;

    v8f acc[2][2] = {};

    // --- A staging: thread -> (row, 16-half chunk) of the 128x32 tile
    const int arow  = threadIdx.x >> 1;
    const int acolh = (threadIdx.x & 1) * 16;
    const _Float16* gA = col + (size_t)(m_base + arow) * KDIM + acolh;
    _Float16* lA0 = &ldsA[0][arow * LDA + acolh];
    _Float16* lA1 = &ldsA[1][arow * LDA + acolh];

    // --- fragment read indices (ISA 16-bit A 16x32 layout)
    const int frow = lane & 15;
    const int kq0  = (lane < 16) ? 0 : 8;    // first 8-half chunk
    const int kq1  = kq0 + 16;               // second 8-half chunk

    // prologue: stage k-tile 0
    *(uint4*)(lA0)     = *(const uint4*)(gA);
    *(uint4*)(lA0 + 8) = *(const uint4*)(gA + 8);
    gA += 32;

    for (int kt = 0; kt < KT; ++kt) {
        __syncthreads();
        const _Float16* lcur = ldsA[kt & 1];
        _Float16*       lnxt = (kt & 1) ? lA0 : lA1;

        // fetch next A tile into registers (overlaps with WMMA below)
        uint4 nx0, nx1;
        if (kt + 1 < KT) {
            nx0 = *(const uint4*)(gA);
            nx1 = *(const uint4*)(gA + 8);
            __builtin_prefetch(gA + 32, 0, 1);   // global_prefetch next-next tile
            gA += 32;
        }

        // load A fragments from LDS
        v16h afrag[2];
#pragma unroll
        for (int mi = 0; mi < 2; ++mi) {
            int r = wy * 32 + mi * 16 + frow;
            v8h lo = *(const v8h*)(lcur + r * LDA + kq0);
            v8h hi = *(const v8h*)(lcur + r * LDA + kq1);
            afrag[mi] = __builtin_shufflevector(lo, hi,
                0, 1, 2, 3, 4, 5, 6, 7, 8, 9, 10, 11, 12, 13, 14, 15);
        }
        // load B fragments (pre-packed, one contiguous 32B read per lane)
        v16h bfrag[2];
#pragma unroll
        for (int ni = 0; ni < 2; ++ni) {
            int nt = (n_base >> 4) + ni;
            const v8h* bp = (const v8h*)(wpack +
                (((size_t)nt * KT + kt) * 32 + lane) * 16);
            bfrag[ni] = __builtin_shufflevector(bp[0], bp[1],
                0, 1, 2, 3, 4, 5, 6, 7, 8, 9, 10, 11, 12, 13, 14, 15);
        }
#pragma unroll
        for (int mi = 0; mi < 2; ++mi)
#pragma unroll
            for (int ni = 0; ni < 2; ++ni)
                acc[mi][ni] = __builtin_amdgcn_wmma_f32_16x16x32_f16(
                    false, afrag[mi], false, bfrag[ni],
                    (short)0, acc[mi][ni], false, false);

        // stage next tile into the other LDS buffer
        if (kt + 1 < KT) {
            *(uint4*)(lnxt)     = nx0;
            *(uint4*)(lnxt + 8) = nx1;
        }
    }

    // epilogue: ReLU + store. C/D layout: lane<16 -> M rows r (VGPR r),
    // lane>=16 -> rows r+8; N = lane&15. 8 consecutive M rows per lane are
    // 8 contiguous floats in NCHW output.
#pragma unroll
    for (int mi = 0; mi < 2; ++mi)
#pragma unroll
        for (int ni = 0; ni < 2; ++ni) {
            int m0 = m_base + wy * 32 + mi * 16 + ((lane >= 16) ? 8 : 0);
            int n  = n_base + ni * 16 + (lane & 15);
            int b  = m0 >> 12;
            int hw = m0 & (HW - 1);
            float* op = out + ((size_t)b * COUT + n) * HW + hw;
            v8f a = acc[mi][ni];
            float4 lo = make_float4(fmaxf(a[0], 0.f), fmaxf(a[1], 0.f),
                                    fmaxf(a[2], 0.f), fmaxf(a[3], 0.f));
            float4 hi = make_float4(fmaxf(a[4], 0.f), fmaxf(a[5], 0.f),
                                    fmaxf(a[6], 0.f), fmaxf(a[7], 0.f));
            *(float4*)(op)     = lo;
            *(float4*)(op + 4) = hi;
        }
}

// ---------------------------------------------------------------------------
extern "C" void kernel_launch(void* const* d_in, const int* in_sizes, int n_in,
                              void* d_out, int out_size, void* d_ws, size_t ws_size,
                              hipStream_t stream)
{
    const float* x        = (const float*)d_in[0];  // [4,256,64,64]
    const float* shape    = (const float*)d_in[1];  // [4,4,64,64]
    const float* w_offset = (const float*)d_in[2];  // [72,4]
    const float* w_deform = (const float*)d_in[3];  // [256,256,3,3]
    float* out = (float*)d_out;                     // [4,256,64,64]

    _Float16* col   = (_Float16*)d_ws;                                   // 75.5 MB
    _Float16* wpack = (_Float16*)((char*)d_ws + (size_t)MDIM * KDIM * 2); // 1.2 MB

    dcn_pack_w<<<(16 * KT * 32 * 16 + 255) / 256, 256, 0, stream>>>(w_deform, wpack);
    dcn_im2col<<<(MDIM * DG * KK9 + 255) / 256, 256, 0, stream>>>(x, shape, w_offset, col);
    dcn_wmma_gemm<<<dim3(MDIM / 128, COUT / 64), 256, 0, stream>>>(col, wpack, out);
}
